// G2TotalLoss_10247791968550
// MI455X (gfx1250) — matrix operand order repaired and checked by reading, not compiled, not run on hardware
//
#include <hip/hip_runtime.h>

typedef __attribute__((ext_vector_type(16))) _Float16 v16h;
typedef __attribute__((ext_vector_type(8)))  float    v8f;
typedef unsigned int u32x4 __attribute__((ext_vector_type(4)));
typedef int          i32x8 __attribute__((ext_vector_type(8)));
typedef int          i32x4 __attribute__((ext_vector_type(4)));

// 3x3 minor determinant of 7x7 matrix G at rows r*, cols c*
__device__ __forceinline__ float det3(const float* G, int r0, int r1, int r2,
                                      int c0, int c1, int c2) {
  float a = G[r0*7+c0], b = G[r0*7+c1], c = G[r0*7+c2];
  float d = G[r1*7+c0], e = G[r1*7+c1], f = G[r1*7+c2];
  float g = G[r2*7+c0], h = G[r2*7+c1], i = G[r2*7+c2];
  return a*(e*i - f*h) + b*(f*g - d*i) + c*(d*h - e*g);
}

// det3 and its directional derivative: dD = sum cof[r][c] * dG[r][c]
__device__ __forceinline__ void det3d(const float* G, const float* dG,
                                      int r0, int r1, int r2,
                                      int c0, int c1, int c2,
                                      float& D, float& dD) {
  float a = G[r0*7+c0], b = G[r0*7+c1], c = G[r0*7+c2];
  float d = G[r1*7+c0], e = G[r1*7+c1], f = G[r1*7+c2];
  float g = G[r2*7+c0], h = G[r2*7+c1], i = G[r2*7+c2];
  float da = dG[r0*7+c0], db = dG[r0*7+c1], dc = dG[r0*7+c2];
  float dd = dG[r1*7+c0], de = dG[r1*7+c1], df = dG[r1*7+c2];
  float dg = dG[r2*7+c0], dh = dG[r2*7+c1], di = dG[r2*7+c2];
  float A0 = e*i - f*h, A1 = f*g - d*i, A2 = d*h - e*g;
  D  = a*A0 + b*A1 + c*A2;
  dD = da*A0 + db*A1 + dc*A2
     + dd*(c*h - b*i) + de*(a*i - c*g) + df*(b*g - a*h)
     + dg*(b*f - c*e) + dh*(c*d - a*f) + di*(a*e - b*d);
}

// One wave (32 lanes) per block; block owns 16 samples (one WMMA row tile).
__global__ __launch_bounds__(32) void g2_loss_kernel(
    const float* __restrict__ coords, const float* __restrict__ W1,
    const float* __restrict__ b1,     const float* __restrict__ Wm,
    const float* __restrict__ Wp,     const int* __restrict__ epochp,
    float* __restrict__ out, int B)
{
  // ---- weights / inputs ----
  __shared__ float    sC [16*7];    // coords tile (TDM destination)
  __shared__ float    sW1[7*64];
  __shared__ float    sb1[64];
  __shared__ _Float16 sWm[64*64];   // 64x49 zero-padded to 64x64 (B operand source)
  __shared__ _Float16 sWp[64*48];   // 64x35 zero-padded to 64x48
  // ---- per-sample state (stride per sample) ----
  __shared__ float sg  [16*49];     // metric
  __shared__ float sgi [16*49];     // inverse metric
  __shared__ float sphi[16*48];     // phi (35 valid)
  __shared__ float spc [16*35];     // raised-index components
  __shared__ float sdphi[16*35];    // dphi accumulator
  __shared__ float sdst[16*21];     // dstar accumulator
  __shared__ float sdiv[16*7];      // divergence accumulator
  __shared__ float sJm [16*64];     // WMMA: raw M / dM (49 valid); later Js
  __shared__ float sJp [16*48];     // WMMA: dphi_k tangent (35 valid)
  __shared__ float sdgi[16*49];     // eig workspace, then dgi
  __shared__ float stmp[16*49];     // dg*gi temp, then dpc
  // ---- combinatorial tables (generated on device) ----
  __shared__ float tHS[35]; __shared__ int tHT[35];
  __shared__ int   tTRI[35*3];
  __shared__ int   tD3C[35*4]; __shared__ int tD3T[35*4]; __shared__ float tD3S[35*4];
  __shared__ int   tD4C[21*5]; __shared__ int tD4T[21*5]; __shared__ float tD4S[21*5];

  const int lane      = threadIdx.x;
  const int row       = lane & 15;      // sample row (A/D) or column (B/D)
  const int grp       = lane >> 4;      // lane half
  const int blockBase = blockIdx.x * 16;
  const bool valid    = (blockBase + row) < B;

  // ------------- coords tile via Tensor Data Mover (async DMA -> LDS) -------------
#if __has_builtin(__builtin_amdgcn_tensor_load_to_lds)
  {
    unsigned lds_addr = (unsigned)(unsigned long long)(void*)sC;   // LDS byte offset
    unsigned long long ga =
        (unsigned long long)(const void*)(coords + (size_t)blockBase * 7);
    u32x4 g0;
    g0[0] = 1u;                                          // count=1 (valid descriptor)
    g0[1] = lds_addr;                                    // lds_addr [63:32]
    g0[2] = (unsigned)(ga & 0xFFFFFFFFu);                // global_addr lo
    g0[3] = (unsigned)((ga >> 32) & 0x01FFFFFFu)         // global_addr hi (57-bit)
          | (2u << 30);                                  // type=2 ("image")
    i32x8 g1;
    g1[0] = 0x00020000;                                  // data_size=2 -> 4 bytes
    g1[1] = (7 << 16);                                   // tensor_dim0 = 7
    g1[2] = (int)(((unsigned)B & 0xFFFFu) << 16);        // tensor_dim1 lo16 = B
    g1[3] = (int)((((unsigned)B >> 16) & 0xFFFFu)        // tensor_dim1 hi16
          | (7u << 16));                                 // tile_dim0 = 7
    g1[4] = 16;                                          // tile_dim1 = 16
    g1[5] = 7;                                           // tensor_dim0_stride = 7
    g1[6] = 0; g1[7] = 0;
    i32x4 z4 = {0, 0, 0, 0};
    i32x8 z8 = {0, 0, 0, 0, 0, 0, 0, 0};
    // clang-23 / therock-10.0 form: 6 args (g0, g1, g2, g3, extra, cpol)
    __builtin_amdgcn_tensor_load_to_lds(g0, g1, z4, z4, z8, 0);
    __builtin_amdgcn_s_wait_tensorcnt(0);                // OOB rows read as zero
  }
#else
  for (int i = lane; i < 16*7; i += 32) {
    int s = i / 7;
    sC[i] = ((blockBase + s) < B) ? coords[(size_t)blockBase*7 + i] : 0.0f;
  }
#endif

  // ---------------- table generation (lane 0) ----------------
  if (lane == 0) {
    int c3[35][3], c4[35][4], c5[21][5];
    int n = 0;
    for (int a = 0; a < 7; a++) for (int b = a+1; b < 7; b++) for (int c = b+1; c < 7; c++) {
      c3[n][0] = a; c3[n][1] = b; c3[n][2] = c;
      tTRI[n*3+0] = a; tTRI[n*3+1] = b; tTRI[n*3+2] = c; n++;
    }
    n = 0;
    for (int a = 0; a < 7; a++) for (int b = a+1; b < 7; b++) for (int c = b+1; c < 7; c++)
      for (int d = c+1; d < 7; d++) { c4[n][0]=a; c4[n][1]=b; c4[n][2]=c; c4[n][3]=d; n++; }
    n = 0;
    for (int a = 0; a < 7; a++) for (int b = a+1; b < 7; b++) for (int c = b+1; c < 7; c++)
      for (int d = c+1; d < 7; d++) for (int e = d+1; e < 7; e++) {
        c5[n][0]=a; c5[n][1]=b; c5[n][2]=c; c5[n][3]=d; c5[n][4]=e; n++;
      }
    auto idx3 = [&](int x, int y, int z) -> int {
      for (int t = 0; t < 35; t++)
        if (c3[t][0]==x && c3[t][1]==y && c3[t][2]==z) return t;
      return 0;
    };
    auto idx4 = [&](const int* r) -> int {
      for (int t = 0; t < 35; t++)
        if (c4[t][0]==r[0] && c4[t][1]==r[1] && c4[t][2]==r[2] && c4[t][3]==r[3]) return t;
      return 0;
    };
    for (int m = 0; m < 35; m++) {       // Hodge sign/target
      bool inq[7] = {false,false,false,false,false,false,false};
      for (int x = 0; x < 4; x++) inq[c4[m][x]] = true;
      int comp[3]; int nc = 0;
      for (int i = 0; i < 7; i++) if (!inq[i]) comp[nc++] = i;
      int perm[7] = {comp[0],comp[1],comp[2],c4[m][0],c4[m][1],c4[m][2],c4[m][3]};
      int invc = 0;
      for (int i = 0; i < 7; i++) for (int j = i+1; j < 7; j++) if (perm[i] > perm[j]) invc++;
      tHS[m] = (invc & 1) ? -1.0f : 1.0f;
      tHT[m] = idx3(comp[0], comp[1], comp[2]);
    }
    for (int m = 0; m < 35; m++) for (int s = 0; s < 4; s++) {  // d(phi) stencil
      int rest[3]; int nr = 0;
      for (int x = 0; x < 4; x++) if (x != s) rest[nr++] = c4[m][x];
      tD3C[m*4+s] = c4[m][s];
      tD3T[m*4+s] = idx3(rest[0], rest[1], rest[2]);
      tD3S[m*4+s] = (s & 1) ? -1.0f : 1.0f;
    }
    for (int m = 0; m < 21; m++) for (int s = 0; s < 5; s++) {  // d(star phi) stencil
      int rest[4]; int nr = 0;
      for (int x = 0; x < 5; x++) if (x != s) rest[nr++] = c5[m][x];
      tD4C[m*5+s] = c5[m][s];
      tD4T[m*5+s] = idx4(rest);
      tD4S[m*5+s] = (s & 1) ? -1.0f : 1.0f;
    }
  }

  // ---------------- weights to LDS ----------------
  for (int i = lane; i < 7*64; i += 32) sW1[i] = W1[i];
  for (int i = lane; i < 64;   i += 32) sb1[i] = b1[i];
  for (int i = lane; i < 64*64; i += 32) {
    int r = i >> 6, c = i & 63;
    sWm[i] = (c < 49) ? (_Float16)Wm[r*49 + c] : (_Float16)0.0f;
  }
  for (int i = lane; i < 64*48; i += 32) {
    int r = i / 48, c = i % 48;
    sWp[i] = (c < 35) ? (_Float16)Wp[r*35 + c] : (_Float16)0.0f;
  }
  __syncthreads();

  // A-operand element -> K mapping (ISA dense 16-bit A 16x32 table):
  //   e<8 -> K = 8*grp + e ; e>=8 -> K = 8*grp + e + 8
  auto kmap = [&](int e) -> int { return 8*grp + e + ((e >= 8) ? 8 : 0); };

  // ---- hoisted B operands: Wm (4 tiles) and Wp (3 tiles), K in 2 chunks of 32 ----
  // B 32x16 striping: same K order as A -> Klocal = kmap(e)
  v16h bWm[8], bWp[6];
  #pragma unroll
  for (int t = 0; t < 4; t++)
    #pragma unroll
    for (int kc = 0; kc < 2; kc++)
      #pragma unroll
      for (int e = 0; e < 16; e++)
        bWm[t*2+kc][e] = sWm[(32*kc + kmap(e))*64 + 16*t + row];
  #pragma unroll
  for (int t = 0; t < 3; t++)
    #pragma unroll
    for (int kc = 0; kc < 2; kc++)
      #pragma unroll
      for (int e = 0; e < 16; e++)
        bWp[t*2+kc][e] = sWp[(32*kc + kmap(e))*48 + 16*t + row];

  // ---- stage A: each lane computes exactly its 32 A-operand entries of h ----
  float uA[32];           // 1 - h^2 at this lane's (row, j) positions
  v16h hA[2];
  #pragma unroll
  for (int kc = 0; kc < 2; kc++)
    #pragma unroll
    for (int e = 0; e < 16; e++) {
      int j = 32*kc + kmap(e);
      float z = sb1[j];
      #pragma unroll
      for (int i = 0; i < 7; i++) z += sC[row*7 + i] * sW1[i*64 + j];
      float h = tanhf(z);
      hA[kc][e] = (_Float16)h;
      uA[kc*16 + e] = 1.0f - h*h;
    }

  // 16xN = (16x64) @ (64xN) via v_wmma_f32_16x16x32_f16
  auto gemm_tiles = [&](const v16h* Bops, int ncol, int ntile,
                        float* dst, const v16h* A) {
    for (int t = 0; t < ntile; t++) {
      v8f acc = {0.f, 0.f, 0.f, 0.f, 0.f, 0.f, 0.f, 0.f};
      #pragma unroll
      for (int kc = 0; kc < 2; kc++)
        acc = __builtin_amdgcn_wmma_f32_16x16x32_f16(
                  false, A[kc], false, Bops[t*2+kc], (short)0, acc, false, false);
      #pragma unroll
      for (int r = 0; r < 8; r++) {
        int m = grp*8 + r;          // C/D: lanes0-15 -> M=r, lanes16-31 -> M=8+r
        dst[m*ncol + 16*t + row] = acc[r];
      }
    }
  };

  // ---------------- stage B: M = H@Wm, phi = H@Wp ----------------
  gemm_tiles(bWm, 64, 4, sJm, hA);
  gemm_tiles(bWp, 48, 3, sphi, hA);
  __syncthreads();

  // ---------------- once-per-sample scalar stage (lanes 0..15) ----------------
  float det = 1.0f, sdet = 0.0f, lossp = 0.0f, pn = 0.0f;
  if (lane < 16) {
    float* g    = sg   + row*49;
    float* gi   = sgi  + row*49;
    float* Mr   = sJm  + row*64;
    float* E    = sdgi + row*49;   // eig workspace
    float* phiA = sphi + row*48;
    float* pc   = spc  + row*35;
    // metric = 0.5(M + M^T) + I
    for (int i = 0; i < 7; i++)
      for (int j = 0; j < 7; j++)
        g[i*7+j] = 0.5f*(Mr[i*7+j] + Mr[j*7+i]) + (i == j ? 1.0f : 0.0f);
    // eigenvalues via cyclic Jacobi
    for (int i = 0; i < 49; i++) E[i] = g[i];
    for (int sw = 0; sw < 6; sw++)
      for (int p = 0; p < 6; p++)
        for (int q = p+1; q < 7; q++) {
          float apq = E[p*7+q];
          if (fabsf(apq) > 1e-12f) {
            float th = (E[q*7+q] - E[p*7+p]) / (2.0f*apq);
            float t  = (th >= 0.0f ? 1.0f : -1.0f) / (fabsf(th) + sqrtf(th*th + 1.0f));
            float cc = 1.0f / sqrtf(t*t + 1.0f);
            float ss = t*cc;
            for (int r = 0; r < 7; r++) {
              float a = E[r*7+p], b = E[r*7+q];
              E[r*7+p] = cc*a - ss*b; E[r*7+q] = ss*a + cc*b;
            }
            for (int r = 0; r < 7; r++) {
              float a = E[p*7+r], b = E[q*7+r];
              E[p*7+r] = cc*a - ss*b; E[q*7+r] = ss*a + cc*b;
            }
          }
        }
    for (int i = 0; i < 7; i++) {
      float rsl = 1e-6f - E[i*7+i];
      if (rsl > 0.0f) lossp += rsl*rsl;
    }
    // in-place Gauss-Jordan inverse + det
    for (int i = 0; i < 49; i++) gi[i] = g[i];
    det = 1.0f;
    for (int kk = 0; kk < 7; kk++) {
      float piv = gi[kk*7+kk]; det *= piv;
      float ip = 1.0f / piv;
      gi[kk*7+kk] = ip;
      for (int j = 0; j < 7; j++) if (j != kk) gi[kk*7+j] *= ip;
      for (int i = 0; i < 7; i++) if (i != kk) {
        float f = gi[i*7+kk];
        for (int j = 0; j < 7; j++) if (j != kk) gi[i*7+j] -= f*gi[kk*7+j];
        gi[i*7+kk] = -f*ip;
      }
    }
    sdet = sqrtf(fmaxf(det, 1e-30f));
    // phi norm
    for (int a = 0; a < 35; a++) pn += phiA[a]*phiA[a];
    // pc[m] = sum_a phi[a] * det3(gi | rows(m), cols(a))
    for (int m = 0; m < 35; m++) {
      int i0 = tTRI[m*3], i1 = tTRI[m*3+1], i2 = tTRI[m*3+2];
      float acc = 0.0f;
      for (int a = 0; a < 35; a++)
        acc += phiA[a]*det3(gi, i0, i1, i2, tTRI[a*3], tTRI[a*3+1], tTRI[a*3+2]);
      pc[m] = acc;
    }
    for (int m = 0; m < 35; m++) sdphi[row*35+m] = 0.0f;
    for (int m = 0; m < 21; m++) sdst[row*21+m]  = 0.0f;
    for (int i = 0; i < 7;  i++) sdiv[row*7+i]   = 0.0f;
  }

  // ---------------- direction loop (forward-mode tangents) ----------------
  for (int k = 0; k < 7; k++) {
    __syncthreads();
    // dh_k = (1-h^2) * W1[k,:]  (A operand, f16) -- uA is register-resident
    v16h dA[2];
    #pragma unroll
    for (int kc = 0; kc < 2; kc++)
      #pragma unroll
      for (int e = 0; e < 16; e++) {
        int j = 32*kc + kmap(e);
        dA[kc][e] = (_Float16)(uA[kc*16 + e] * sW1[k*64 + j]);
      }
    gemm_tiles(bWm, 64, 4, sJm, dA);   // dM_k
    gemm_tiles(bWp, 48, 3, sJp, dA);   // dphi_k
    __syncthreads();

    if (lane < 16) {
      float* Jm   = sJm  + row*64;
      float* Jp   = sJp  + row*48;
      float* gi   = sgi  + row*49;
      float* dgi  = sdgi + row*49;
      float* tmp  = stmp + row*49;
      float* phiA = sphi + row*48;
      float* pc   = spc  + row*35;
      // dg = 0.5(dM + dM^T), in place
      for (int i = 0; i < 7; i++)
        for (int j = i; j < 7; j++) {
          float v = 0.5f*(Jm[i*7+j] + Jm[j*7+i]);
          Jm[i*7+j] = v; Jm[j*7+i] = v;
        }
      // div_i += dg[i][k]
      for (int i = 0; i < 7; i++) sdiv[row*7+i] += Jm[i*7+k];
      // ds = 0.5 * sqrt(det) * tr(gi*dg)
      float tr = 0.0f;
      for (int i = 0; i < 7; i++)
        for (int j = 0; j < 7; j++) tr += gi[i*7+j]*Jm[j*7+i];
      float ds = 0.5f*sdet*tr;
      // dgi = -gi*dg*gi
      for (int i = 0; i < 7; i++)
        for (int j = 0; j < 7; j++) {
          float acc = 0.0f;
          for (int l = 0; l < 7; l++) acc += Jm[i*7+l]*gi[l*7+j];
          tmp[i*7+j] = acc;
        }
      for (int i = 0; i < 7; i++)
        for (int j = 0; j < 7; j++) {
          float acc = 0.0f;
          for (int l = 0; l < 7; l++) acc += gi[i*7+l]*tmp[l*7+j];
          dgi[i*7+j] = -acc;
        }
      // dpc[m] = sum_a dphi[a]*D(m,a) + phi[a]*dD(m,a)   (into tmp[0..34])
      for (int m = 0; m < 35; m++) {
        int i0 = tTRI[m*3], i1 = tTRI[m*3+1], i2 = tTRI[m*3+2];
        float acc = 0.0f;
        for (int a = 0; a < 35; a++) {
          float D, dD;
          det3d(gi, dgi, i0, i1, i2, tTRI[a*3], tTRI[a*3+1], tTRI[a*3+2], D, dD);
          acc += Jp[a]*D + phiA[a]*dD;
        }
        tmp[m] = acc;
      }
      // Js_k[m] = HS[m]*(ds*pc[HT[m]] + s*dpc[HT[m]])   (into Jm[0..34])
      for (int m = 0; m < 35; m++)
        Jm[m] = tHS[m]*(ds*pc[tHT[m]] + sdet*tmp[tHT[m]]);
      // accumulate exterior derivatives for this direction
      for (int m = 0; m < 35; m++)
        for (int s2 = 0; s2 < 4; s2++)
          if (tD3C[m*4+s2] == k)
            sdphi[row*35+m] += tD3S[m*4+s2]*Jp[tD3T[m*4+s2]];
      for (int m = 0; m < 21; m++)
        for (int s2 = 0; s2 < 5; s2++)
          if (tD4C[m*5+s2] == k)
            sdst[row*21+m] += tD4S[m*5+s2]*Jm[tD4T[m*5+s2]];
    }
  }
  __syncthreads();

  // ---------------- loss assembly + reduction ----------------
  int epoch = *epochp;
  float wt, wv, wn, wg;
  if      (epoch <  500) { wt = 0.1f;  wv = 10.0f; wn = 1.0f; wg = 0.1f; }
  else if (epoch < 2000) { wt = 1.0f;  wv = 1.0f;  wn = 1.0f; wg = 0.1f; }
  else                   { wt = 10.0f; wv = 0.1f;  wn = 1.0f; wg = 0.1f; }

  float sampleLoss = 0.0f;
  if (lane < 16 && valid) {
    float lt = 0.0f;
    for (int m = 0; m < 35; m++) { float v = sdphi[row*35+m]; lt += v*v; }
    for (int m = 0; m < 21; m++) { float v = sdst[row*21+m];  lt += v*v; }
    float lg = 0.0f;
    for (int i = 0; i < 7; i++) { float v = sdiv[row*7+i]; lg += v*v; }
    float dv = det - 1.0f, dn = pn - 7.0f;
    sampleLoss = wt*lt + wv*dv*dv + wn*dn*dn
               + wg*lg*(1.0f/7.0f) + lossp*(1.0f/7.0f);
  }
  if (lane < 16) stmp[lane] = sampleLoss;
  __syncthreads();
  if (lane == 0) {
    float bs = 0.0f;
    for (int i = 0; i < 16; i++) bs += stmp[i];
    atomicAdd(out, bs / (float)B);
  }
}

extern "C" void kernel_launch(void* const* d_in, const int* in_sizes, int n_in,
                              void* d_out, int out_size, void* d_ws, size_t ws_size,
                              hipStream_t stream) {
  const float* coords = (const float*)d_in[0];
  const float* W1     = (const float*)d_in[1];
  const float* b1     = (const float*)d_in[2];
  const float* Wm     = (const float*)d_in[3];
  const float* Wp     = (const float*)d_in[4];
  const int*   epoch  = (const int*)d_in[5];
  float* out = (float*)d_out;
  int B = in_sizes[0] / 7;
  (void)hipMemsetAsync(out, 0, sizeof(float), stream);
  int blocks = (B + 15) / 16;
  g2_loss_kernel<<<blocks, 32, 0, stream>>>(coords, W1, b1, Wm, Wp, epoch, out, B);
}
